// EnhancedPEFTWindowAttention_12850542150466
// MI455X (gfx1250) — compile-verified, matrix operand order checked by reading, not run on hardware
//
#include <hip/hip_runtime.h>

typedef __attribute__((ext_vector_type(16))) __bf16 bf16x16;
typedef __attribute__((ext_vector_type(8)))  float  f32x8;

#define N_TOK   49
#define H_      12
#define D_      384
#define HD_     32
#define B_TOT   2048
#define M_ROWS  (B_TOT * N_TOK)     /* 100352 */
#define R_      16
#define SCALING_ 2.0f
#define SCALE_   0.17677669529663687f   /* 32^-0.5 */

#define WMMA_BF16(a, b, c) \
  __builtin_amdgcn_wmma_f32_16x16x32_bf16(false, (a), false, (b), (short)0, (c), false, false)

// ---- CDNA5 async LDS DMA (ASYNCcnt path, ISA ch.10 / 15.18) ----
// Generic pointers to __shared__ carry the LDS byte offset in the low 32 bits
// (LDS aperture); hardware adds LDS_BASE.
static __device__ inline unsigned lds_off32(const void* p) {
  return (unsigned)(unsigned long long)p;
}
static __device__ inline void async_load_b128(unsigned lds_off, const void* gptr) {
  asm volatile("global_load_async_to_lds_b128 %0, %1, off"
               :: "v"(lds_off), "v"((unsigned long long)gptr) : "memory");
}
static __device__ inline void wait_async0() {
  asm volatile("s_wait_asynccnt 0" ::: "memory");
}

// ---- fragment loaders (ISA 7.12.2 16-bit layouts, wave32) ----
// Per lane the A/B fragment is two contiguous 16B chunks of one row:
// u32[0..3] = row[4g .. 4g+3], u32[4..7] = row[8+4g .. 11+4g]  -> two b128 loads.
static __device__ inline bf16x16 frag_rowmajor(const __bf16* base, int stride_halfs,
                                               int row_base, int lane) {
  int r = row_base + (lane & 15);
  int g = lane >> 4;
  const uint4* p = (const uint4*)(base + (size_t)r * stride_halfs);
  union { bf16x16 v; uint4 q[2]; } f;
  f.q[0] = p[g];
  f.q[1] = p[2 + g];
  return f.v;
}

static __device__ inline bf16x16 frag_colmajor(const __bf16* base, int stride_halfs,
                                               int col_base, int kbase, int lane) {
  int c = col_base + (lane & 15);
  int g = lane >> 4;
  bf16x16 f;
#pragma unroll
  for (int j = 0; j < 8; ++j) {
    int k0 = (j < 4 ? 2 * j : 16 + 2 * (j - 4)) + 8 * g + kbase;
    f[2 * j]     = base[(k0)     * stride_halfs + c];
    f[2 * j + 1] = base[(k0 + 1) * stride_halfs + c];
  }
  return f;
}

// ---- prep kernels: fp32 -> bf16 (optionally fused add for W + Res folding) ----
__global__ void cvt_bf16_kernel(const float* __restrict__ a, __bf16* __restrict__ o, int n) {
  int i = blockIdx.x * 256 + threadIdx.x;
  if (i < n) o[i] = (__bf16)a[i];
}
__global__ void cvt_add_bf16_kernel(const float* __restrict__ a, const float* __restrict__ b,
                                    __bf16* __restrict__ o, int n) {
  int i = blockIdx.x * 256 + threadIdx.x;
  if (i < n) o[i] = (__bf16)(a[i] + b[i]);
}

// ---- LoRA down-projection: xd[M,16] = x[M,384] @ down[16,384]^T ----
__global__ __launch_bounds__(128) void lora_down_kernel(const __bf16* __restrict__ xb,
                                                        const __bf16* __restrict__ dn,
                                                        float* __restrict__ xd) {
  __shared__ __attribute__((aligned(16))) __bf16 ldn[16 * 384];
  const int tid = threadIdx.x, lane = tid & 31, w = tid >> 5;
  unsigned int* s = (unsigned int*)ldn;
  const unsigned int* gd = (const unsigned int*)dn;
#pragma unroll
  for (int i = 0; i < 24; ++i) s[i * 128 + tid] = gd[i * 128 + tid];
  __syncthreads();
  const int m0 = (blockIdx.x * 4 + w) * 16;
  f32x8 acc = {};
  for (int kb = 0; kb < 384; kb += 32) {
    bf16x16 a = frag_rowmajor(xb + kb, 384, m0, lane);
    bf16x16 b = frag_rowmajor(ldn + kb, 384, 0, lane);
    acc = WMMA_BF16(a, b, acc);
  }
  const int g = lane >> 4, li = lane & 15;
#pragma unroll
  for (int r = 0; r < 8; ++r) {
    int row = m0 + r + 8 * g;
    xd[(size_t)row * 16 + li] = acc[r];
  }
}

// ---- fused gated-LoRA linear with async double-buffered K-loop:
// out = x@(W+Res)^T + b + sigmoid(x@G^T) * SCALING * (xd @ Up^T)
template <typename OutT>
__global__ __launch_bounds__(256) void lora_gemm_kernel(
    const __bf16* __restrict__ xb, const __bf16* __restrict__ wc,
    const __bf16* __restrict__ wg, const float* __restrict__ bias,
    const float* __restrict__ up, const float* __restrict__ xd,
    OutT* __restrict__ out, int nout) {
  __shared__ __attribute__((aligned(16))) __bf16 lx[2][128 * 32];
  __shared__ __attribute__((aligned(16))) __bf16 lwc[2][64 * 32];
  __shared__ __attribute__((aligned(16))) __bf16 lwg[2][64 * 32];
  const int tid = threadIdx.x, lane = tid & 31, w = tid >> 5;
  const int m0 = blockIdx.x * 128;
  const int n0 = blockIdx.y * 64;

  // Issue one K-tile's worth of async b128 DMA into LDS buffer `bf`.
  // X tile: 128 rows x 64B = 512 chunks (2/thread); W tiles: 256 chunks (1/thread each).
  auto issue_tile = [&](int kb, int bf) {
    const char* gx = (const char*)xb;
    const char* gc = (const char*)wc;
    const char* gg = (const char*)wg;
    unsigned bx = lds_off32(&lx[bf][0]);
    unsigned bc = lds_off32(&lwc[bf][0]);
    unsigned bg = lds_off32(&lwg[bf][0]);
#pragma unroll
    for (int i = 0; i < 2; ++i) {
      int c = tid + i * 256;                 // 0..511
      int row = c >> 2, cq = c & 3;
      async_load_b128(bx + c * 16,
                      gx + ((size_t)(m0 + row) * 384 + kb) * 2 + cq * 16);
    }
    {
      int row = tid >> 2, cq = tid & 3;
      size_t gb = ((size_t)(n0 + row) * 384 + kb) * 2 + cq * 16;
      async_load_b128(bc + tid * 16, gc + gb);
      async_load_b128(bg + tid * 16, gg + gb);
    }
  };

  f32x8 accC[4] = {};
  f32x8 accG[4] = {};

  issue_tile(0, 0);
  int buf = 0;
  for (int kb = 0; kb < 384; kb += 32) {
    wait_async0();        // own wave's DMA done
    __syncthreads();      // everyone's DMA visible; prev buffer free for reuse
    if (kb + 32 < 384) issue_tile(kb + 32, buf ^ 1);

    bf16x16 a = frag_rowmajor(&lx[buf][0], 32, w * 16, lane);
#pragma unroll
    for (int t = 0; t < 4; ++t) {
      bf16x16 bc = frag_rowmajor(&lwc[buf][0], 32, t * 16, lane);
      accC[t] = WMMA_BF16(a, bc, accC[t]);
      bf16x16 bg = frag_rowmajor(&lwg[buf][0], 32, t * 16, lane);
      accG[t] = WMMA_BF16(a, bg, accG[t]);
    }
    buf ^= 1;
  }

  const int g = lane >> 4, li = lane & 15;
#pragma unroll
  for (int t = 0; t < 4; ++t) {
    int col = n0 + t * 16 + li;
    float bcol = bias[col];
    float upc[16];
#pragma unroll
    for (int rr = 0; rr < 16; ++rr) upc[rr] = up[(size_t)col * 16 + rr];
#pragma unroll
    for (int r = 0; r < 8; ++r) {
      int row = m0 + w * 16 + r + 8 * g;
      const float* xdr = xd + (size_t)row * 16;
      float l = 0.f;
#pragma unroll
      for (int rr = 0; rr < 16; ++rr) l += xdr[rr] * upc[rr];
      float gt = 1.f / (1.f + __expf(-accG[t][r]));
      float val = accC[t][r] + bcol + gt * l * SCALING_;
      out[(size_t)row * nout + col] = (OutT)val;
    }
  }
}

// ---- windowed attention, one block per (head, window). N padded 49 -> 64. ----
__global__ __launch_bounds__(128) void attn_kernel(
    const __bf16* __restrict__ qkvb, const float* __restrict__ bias_table,
    const int* __restrict__ rel_index, const float* __restrict__ mask,
    __bf16* __restrict__ xa) {
  __shared__ __attribute__((aligned(16))) __bf16 lq[64 * 32];
  __shared__ __attribute__((aligned(16))) __bf16 lk[64 * 32];
  __shared__ __attribute__((aligned(16))) __bf16 lv[64 * 32];
  __shared__ __attribute__((aligned(16))) float  ls[64 * 64];
  __shared__ __attribute__((aligned(16))) __bf16 lp[64 * 64];
  const int h = blockIdx.x;
  const int b = blockIdx.y;
  const int tid = threadIdx.x, lane = tid & 31, w = tid >> 5;

  // Zero the 49..63 padding rows of q/k/v via DS (disjoint from async targets).
  {
    unsigned int* sq = (unsigned int*)lq;
    unsigned int* sk = (unsigned int*)lk;
    unsigned int* sv = (unsigned int*)lv;
    for (int idx = tid; idx < 720; idx += 128) {  // 3 mats * 15 rows * 16 u32
      int mat = idx / 240, rem = idx % 240;
      int off = (N_TOK * 16) + rem;               // rows 49..63
      (mat == 0 ? sq : (mat == 1 ? sk : sv))[off] = 0u;
    }
  }
  // Async DMA the 49 valid rows of q/k/v (3*49*4 = 588 b128 chunks).
  {
    const char* gq = (const char*)qkvb;
    unsigned base[3] = { lds_off32(lq), lds_off32(lk), lds_off32(lv) };
    for (int idx = tid; idx < 588; idx += 128) {
      int mat = idx / 196, rem = idx % 196;
      int row = rem >> 2, cq = rem & 3;
      size_t gb = ((size_t)(b * N_TOK + row) * (3 * D_) + mat * D_ + h * HD_) * 2 + cq * 16;
      async_load_b128(base[mat] + row * 64 + cq * 16, gq + gb);
    }
  }
  wait_async0();
  __syncthreads();

  // S = q @ k^T  (each wave: one 16-row strip, 4 column tiles, K=32 in one WMMA)
  {
    const int g = lane >> 4, li = lane & 15;
    bf16x16 a = frag_rowmajor(lq, 32, w * 16, lane);
#pragma unroll
    for (int t = 0; t < 4; ++t) {
      bf16x16 bk = frag_rowmajor(lk, 32, t * 16, lane);
      f32x8 s = {};
      s = WMMA_BF16(a, bk, s);
#pragma unroll
      for (int r = 0; r < 8; ++r)
        ls[(w * 16 + r + 8 * g) * 64 + t * 16 + li] = s[r];
    }
  }
  __syncthreads();

  // fused scale + rel-pos-bias gather + window mask + softmax (one row per lane)
  if (tid < 64) {
    int row = tid;
    if (row < N_TOK) {
      float mx = -1e30f;
      for (int m = 0; m < N_TOK; ++m) {
        int rel = rel_index[row * N_TOK + m];
        float v = ls[row * 64 + m] * SCALE_ + bias_table[rel * H_ + h] +
                  mask[(size_t)(b & 63) * (N_TOK * N_TOK) + row * N_TOK + m];
        ls[row * 64 + m] = v;
        mx = fmaxf(mx, v);
      }
      float sum = 0.f;
      for (int m = 0; m < N_TOK; ++m) {
        float e = __expf(ls[row * 64 + m] - mx);
        ls[row * 64 + m] = e;
        sum += e;
      }
      float inv = 1.f / sum;
      for (int m = 0; m < N_TOK; ++m) lp[row * 64 + m] = (__bf16)(ls[row * 64 + m] * inv);
      for (int m = N_TOK; m < 64; ++m) lp[row * 64 + m] = (__bf16)0.f;
    } else {
      for (int m = 0; m < 64; ++m) lp[row * 64 + m] = (__bf16)0.f;
    }
  }
  __syncthreads();

  // O = P @ V  (K=64 -> two 32-wide steps; 2 column tiles of 16)
  {
    const int g = lane >> 4, li = lane & 15;
    bf16x16 a0 = frag_rowmajor(lp, 64, w * 16, lane);
    bf16x16 a1 = frag_rowmajor(lp + 32, 64, w * 16, lane);
#pragma unroll
    for (int t = 0; t < 2; ++t) {
      bf16x16 b0 = frag_colmajor(lv, 32, t * 16, 0, lane);
      bf16x16 b1 = frag_colmajor(lv, 32, t * 16, 32, lane);
      f32x8 o = {};
      o = WMMA_BF16(a0, b0, o);
      o = WMMA_BF16(a1, b1, o);
#pragma unroll
      for (int r = 0; r < 8; ++r) {
        int row = w * 16 + r + 8 * g;
        if (row < N_TOK)
          xa[(size_t)(b * N_TOK + row) * D_ + h * HD_ + t * 16 + li] = (__bf16)o[r];
      }
    }
  }
}

extern "C" void kernel_launch(void* const* d_in, const int* in_sizes, int n_in,
                              void* d_out, int out_size, void* d_ws, size_t ws_size,
                              hipStream_t stream) {
  const float* x          = (const float*)d_in[0];
  const float* mask       = (const float*)d_in[1];
  const float* qkv_w      = (const float*)d_in[2];
  const float* qkv_b      = (const float*)d_in[3];
  const float* qkv_down   = (const float*)d_in[4];
  const float* qkv_up     = (const float*)d_in[5];
  const float* qkv_gate   = (const float*)d_in[6];
  const float* qkv_res    = (const float*)d_in[7];
  const float* proj_w     = (const float*)d_in[8];
  const float* proj_b     = (const float*)d_in[9];
  const float* proj_down  = (const float*)d_in[10];
  const float* proj_up    = (const float*)d_in[11];
  const float* proj_gate  = (const float*)d_in[12];
  const float* proj_res   = (const float*)d_in[13];
  const float* bias_table = (const float*)d_in[14];
  const int*   rel_index  = (const int*)d_in[15];

  char* ws = (char*)d_ws;
  size_t off = 0;
  auto alloc = [&](size_t bytes) -> char* {
    char* p = ws + off;
    off = (off + bytes + 255) & ~(size_t)255;
    return p;
  };

  __bf16* xb    = (__bf16*)alloc((size_t)M_ROWS * D_ * 2);
  __bf16* qkvb  = (__bf16*)alloc((size_t)M_ROWS * 3 * D_ * 2);
  __bf16* xa    = (__bf16*)alloc((size_t)M_ROWS * D_ * 2);
  float*  xd1   = (float*)alloc((size_t)M_ROWS * R_ * 4);
  float*  xd2   = (float*)alloc((size_t)M_ROWS * R_ * 4);
  __bf16* wc_q  = (__bf16*)alloc((size_t)3 * D_ * D_ * 2);
  __bf16* wg_q  = (__bf16*)alloc((size_t)3 * D_ * D_ * 2);
  __bf16* dn_q  = (__bf16*)alloc((size_t)R_ * D_ * 2);
  __bf16* wc_p  = (__bf16*)alloc((size_t)D_ * D_ * 2);
  __bf16* wg_p  = (__bf16*)alloc((size_t)D_ * D_ * 2);
  __bf16* dn_p  = (__bf16*)alloc((size_t)R_ * D_ * 2);

  auto blocks = [](int n) { return (n + 255) / 256; };
  const int nQ = 3 * D_ * D_, nP = D_ * D_, nD = R_ * D_, nX = M_ROWS * D_;

  // weight prep (fold Res into W; bf16 everything that feeds WMMA)
  cvt_add_bf16_kernel<<<blocks(nQ), 256, 0, stream>>>(qkv_w, qkv_res, wc_q, nQ);
  cvt_bf16_kernel    <<<blocks(nQ), 256, 0, stream>>>(qkv_gate, wg_q, nQ);
  cvt_bf16_kernel    <<<blocks(nD), 256, 0, stream>>>(qkv_down, dn_q, nD);
  cvt_add_bf16_kernel<<<blocks(nP), 256, 0, stream>>>(proj_w, proj_res, wc_p, nP);
  cvt_bf16_kernel    <<<blocks(nP), 256, 0, stream>>>(proj_gate, wg_p, nP);
  cvt_bf16_kernel    <<<blocks(nD), 256, 0, stream>>>(proj_down, dn_p, nD);
  cvt_bf16_kernel    <<<blocks(nX), 256, 0, stream>>>(x, xb, nX);

  // QKV: down-proj, then fused gated-LoRA GEMM (bf16 out)
  lora_down_kernel<<<M_ROWS / 64, 128, 0, stream>>>(xb, dn_q, xd1);
  lora_gemm_kernel<__bf16><<<dim3(M_ROWS / 128, (3 * D_) / 64), 256, 0, stream>>>(
      xb, wc_q, wg_q, qkv_b, qkv_up, xd1, qkvb, 3 * D_);

  // attention per (head, window)
  attn_kernel<<<dim3(H_, B_TOT), 128, 0, stream>>>(qkvb, bias_table, rel_index, mask, xa);

  // proj: down-proj, then fused gated-LoRA GEMM (f32 out to d_out)
  lora_down_kernel<<<M_ROWS / 64, 128, 0, stream>>>(xa, dn_p, xd2);
  lora_gemm_kernel<float><<<dim3(M_ROWS / 128, D_ / 64), 256, 0, stream>>>(
      xa, wc_p, wg_p, proj_b, proj_up, xd2, (float*)d_out, D_);
}
